// ThreeGATCN_87720412053582
// MI455X (gfx1250) — matrix operand order repaired and checked by reading, not compiled, not run on hardware
//
#include <hip/hip_runtime.h>
#include <hip/hip_bf16.h>
#include <stdint.h>

// ---------- types for WMMA ----------
typedef __attribute__((ext_vector_type(16))) _Float16 v16h;
typedef __attribute__((ext_vector_type(8)))  _Float16 v8h;
typedef __attribute__((ext_vector_type(8)))  float    v8f;

#define TM 128
#define TN 64
#define TK 32

// ---------- small utility kernels ----------
__global__ __launch_bounds__(256) void gat_fill_f32(float* __restrict__ p, float v, int n) {
    int i = blockIdx.x * blockDim.x + threadIdx.x;
    if (i < n) p[i] = v;
}

__global__ __launch_bounds__(256) void gat_f32_to_f16(const float* __restrict__ s,
                                                      _Float16* __restrict__ d, int n) {
    int i = blockIdx.x * blockDim.x + threadIdx.x;
    if (i < n) d[i] = (_Float16)s[i];
}

// WT[n*K + k] = (f16) W[k*N + n]   (W is K x N row-major)
__global__ __launch_bounds__(256) void gat_convT(const float* __restrict__ W,
                                                 _Float16* __restrict__ WT, int K, int N) {
    int i = blockIdx.x * blockDim.x + threadIdx.x;
    if (i >= K * N) return;
    int k = i / N;
    int n = i - k * N;
    WT[(size_t)n * K + k] = (_Float16)W[i];
}

// ---------- WMMA GEMM: C[M,N] = A[M,K] (f16) * BT[N,K]^T (f16) + bias ----------
// Block: 256 threads = 8 waves; tile 128(M) x 64(N); K-step 32.
// Wave w computes rows [w*16, w*16+16) x all 64 cols -> 4 wmma accumulators.
// Full-height blocks use GLOBAL_LOAD_ASYNC_TO_LDS_B128 (ASYNCcnt-tracked DMA
// into LDS, no VGPR staging); only the tail M-block takes the guarded path.
__global__ __launch_bounds__(256) void gat_gemm_wmma(const _Float16* __restrict__ A,
                                                     const _Float16* __restrict__ BT,
                                                     const float* __restrict__ bias,
                                                     float* __restrict__ C,
                                                     int M, int N, int K) {
    __shared__ __align__(16) _Float16 sA[TM * TK];  // 8 KB
    __shared__ __align__(16) _Float16 sB[TN * TK];  // 4 KB, stored as [n][k]

    const int m0   = blockIdx.x * TM;
    const int n0   = blockIdx.y * TN;
    const int tid  = threadIdx.x;
    const int lane = tid & 31;
    const int wave = tid >> 5;
    const bool fullM = (m0 + TM) <= M;   // uniform across the block

    // per-thread tile-load coordinates (16B vector granularity)
    const int rA = tid >> 2;             // A rows 0..63 (second half = rA+64)
    const int cA = (tid & 3) << 3;       // halfword column offset 0/8/16/24
    const int rB = tid >> 2;             // B rows 0..63
    const int cB = (tid & 3) << 3;

    // LDS byte addresses (generic-pointer low 32 bits == LDS byte offset)
    const unsigned ldsA0 = (unsigned)(unsigned long long)(const void*)(sA + rA * TK + cA);
    const unsigned ldsA1 = ldsA0 + (unsigned)(64 * TK * 2);  // +64 rows
    const unsigned ldsB  = (unsigned)(unsigned long long)(const void*)(sB + rB * TK + cB);

    v8f acc[4];
    for (int j = 0; j < 4; ++j)
        for (int i = 0; i < 8; ++i) acc[j][i] = 0.0f;

    for (int ko = 0; ko < K; ko += TK) {
        if (fullM) {
            // async DMA: global -> LDS, 3 x 16B per thread, tracked by ASYNCcnt
            unsigned long long gA0 =
                (unsigned long long)(A + (size_t)(m0 + rA) * K + ko + cA);
            unsigned long long gA1 = gA0 + (unsigned long long)64 * K * 2;
            unsigned long long gB =
                (unsigned long long)(BT + (size_t)(n0 + rB) * K + ko + cB);
            asm volatile("global_load_async_to_lds_b128 %0, %1, off"
                         :: "v"(ldsA0), "v"(gA0) : "memory");
            asm volatile("global_load_async_to_lds_b128 %0, %1, off"
                         :: "v"(ldsA1), "v"(gA1) : "memory");
            asm volatile("global_load_async_to_lds_b128 %0, %1, off"
                         :: "v"(ldsB), "v"(gB) : "memory");
            if (ko + TK < K)
                __builtin_prefetch(A + (size_t)(m0 + rA) * K + ko + TK + cA, 0, 1);
            asm volatile("s_wait_asynccnt 0" ::: "memory");
        } else {
            // guarded tail block: synchronous staging with zero padding
            for (int v = tid; v < (TM * TK / 8); v += 256) {
                int r  = v >> 2;
                int c8 = (v & 3) << 3;
                int gr = m0 + r;
                v8h val;
                if (gr < M) {
                    val = *(const v8h*)(A + (size_t)gr * K + ko + c8);
                } else {
                    for (int i = 0; i < 8; ++i) val[i] = (_Float16)0.0f;
                }
                *(v8h*)(sA + r * TK + c8) = val;
            }
            *(v8h*)(sB + rB * TK + cB) =
                *(const v8h*)(BT + (size_t)(n0 + rB) * K + ko + cB);
        }
        __syncthreads();

        // A fragment: 16x32 f16, ISA layout — lanes 0-15 hold K 0..7 / 16..23,
        // lanes 16-31 hold K 8..15 / 24..31, row M = lane%16 (+ wave subtile).
        const int mrow = (wave << 4) + (lane & 15);
        const int kb0  = (lane < 16) ? 0 : 8;
        v8h a0 = *(const v8h*)(sA + mrow * TK + kb0);
        v8h a1 = *(const v8h*)(sA + mrow * TK + kb0 + 16);
        v16h a;
        for (int i = 0; i < 8; ++i) { a[i] = a0[i]; a[i + 8] = a1[i]; }

        for (int j = 0; j < 4; ++j) {
            // B fragment: col N = lane%16, lanes 0-15 K 0..7 / 16..23, lanes 16-31 K 8..15 / 24..31
            const int nrow = (j << 4) + (lane & 15);
            const int kh   = (lane < 16) ? 0 : 8;
            v8h b0 = *(const v8h*)(sB + nrow * TK + kh);
            v8h b1 = *(const v8h*)(sB + nrow * TK + kh + 16);
            v16h b;
            for (int i = 0; i < 8; ++i) { b[i] = b0[i]; b[i + 8] = b1[i]; }
            acc[j] = __builtin_amdgcn_wmma_f32_16x16x32_f16(
                false, a, false, b, (short)0, acc[j], false, false);
        }
        __syncthreads();
    }

    // store: C/D layout — VGPR r: lanes 0-15 M=r, lanes 16-31 M=r+8; col = lane%16
    const int colBase = n0 + (lane & 15);
    const int rowOff  = (lane < 16) ? 0 : 8;
    for (int j = 0; j < 4; ++j) {
        int col  = colBase + (j << 4);
        float bv = bias[col];
        for (int r = 0; r < 8; ++r) {
            int row = m0 + (wave << 4) + rowOff + r;
            if (row < M) C[(size_t)row * N + col] = acc[j][r] + bv;
        }
    }
}

// ---------- edge kernels (wave32 per edge) ----------
__device__ inline void atomicMaxF32(float* addr, float val) {
    if (val >= 0.0f) atomicMax((int*)addr, __float_as_int(val));
    else             atomicMin((unsigned int*)addr, __float_as_uint(val));
}

__global__ __launch_bounds__(256) void gat_edge_logit(
    const float* __restrict__ xl, const float* __restrict__ xr,
    const float* __restrict__ att,
    const long long* __restrict__ srcA, const long long* __restrict__ dstA,
    int E, int Nn, int F, float* __restrict__ elog, float* __restrict__ emax) {
    int wid  = blockIdx.x * (blockDim.x >> 5) + (threadIdx.x >> 5);
    int lane = threadIdx.x & 31;
    int E2   = E + Nn;
    if (wid >= E2) return;
    int s, d;
    if (wid < E) { s = (int)srcA[wid]; d = (int)dstA[wid]; }
    else         { s = wid - E; d = s; }
    const float* pl = xl + (size_t)s * F;
    const float* pr = xr + (size_t)d * F;
    float acc = 0.0f;
    for (int f = lane; f < F; f += 32) {
        float m  = pl[f] + pr[f];
        float lr = m > 0.0f ? m : 0.2f * m;
        acc += lr * att[f];
    }
    for (int off = 16; off > 0; off >>= 1) acc += __shfl_down(acc, off, 32);
    if (lane == 0) {
        elog[wid] = acc;
        atomicMaxF32(emax + d, acc);
    }
}

__global__ __launch_bounds__(256) void gat_edge_exp(
    const float* __restrict__ elog, const float* __restrict__ emax,
    const long long* __restrict__ dstA, int E, int Nn,
    float* __restrict__ ez, float* __restrict__ den) {
    int i = blockIdx.x * blockDim.x + threadIdx.x;
    int E2 = E + Nn;
    if (i >= E2) return;
    int d = (i < E) ? (int)dstA[i] : (i - E);
    float v = __expf(elog[i] - emax[d]);
    ez[i] = v;
    atomicAdd(den + d, v);
}

__global__ __launch_bounds__(256) void gat_edge_aggr(
    const float* __restrict__ ez, const float* __restrict__ den,
    const float* __restrict__ xl,
    const long long* __restrict__ srcA, const long long* __restrict__ dstA,
    int E, int Nn, int F, float* __restrict__ agg) {
    int wid  = blockIdx.x * (blockDim.x >> 5) + (threadIdx.x >> 5);
    int lane = threadIdx.x & 31;
    int E2   = E + Nn;
    if (wid >= E2) return;
    int s, d;
    if (wid < E) { s = (int)srcA[wid]; d = (int)dstA[wid]; }
    else         { s = wid - E; d = s; }
    float alpha = ez[wid] / den[d];
    const float* pl = xl + (size_t)s * F;
    float* po = agg + (size_t)d * F;
    for (int f = lane; f < F; f += 32) atomicAdd(po + f, alpha * pl[f]);
}

__global__ __launch_bounds__(256) void gat_finalize(
    const float* __restrict__ agg, const float* __restrict__ b,
    _Float16* __restrict__ h, int Nn, int F) {
    int i = blockIdx.x * blockDim.x + threadIdx.x;
    if (i >= Nn * F) return;
    float v = agg[i] + b[i % F];
    h[i] = (_Float16)(v > 0.0f ? v : 0.0f);
}

// ---------- orchestration ----------
extern "C" void kernel_launch(void* const* d_in, const int* in_sizes, int n_in,
                              void* d_out, int out_size, void* d_ws, size_t ws_size,
                              hipStream_t stream) {
    const float*     x  = (const float*)d_in[0];
    const long long* ei = (const long long*)d_in[1];
    const int Nn = in_sizes[0] / 128;
    const int E  = in_sizes[1] / 2;
    const int E2 = E + Nn;
    const long long* srcA = ei;
    const long long* dstA = ei + E;

    const float* Wl[3]  = {(const float*)d_in[2],  (const float*)d_in[8],  (const float*)d_in[14]};
    const float* bl[3]  = {(const float*)d_in[3],  (const float*)d_in[9],  (const float*)d_in[15]};
    const float* Wr[3]  = {(const float*)d_in[4],  (const float*)d_in[10], (const float*)d_in[16]};
    const float* br[3]  = {(const float*)d_in[5],  (const float*)d_in[11], (const float*)d_in[17]};
    const float* att[3] = {(const float*)d_in[6],  (const float*)d_in[12], (const float*)d_in[18]};
    const float* bb[3]  = {(const float*)d_in[7],  (const float*)d_in[13], (const float*)d_in[19]};
    const float* Wc = (const float*)d_in[20];
    const float* bc = (const float*)d_in[21];
    const int Fd[4] = {128, 1024, 512, 128};

    // workspace bump allocator
    char* p = (char*)d_ws;
    auto alloc = [&](size_t bytes) -> char* {
        char* r = p;
        p += (bytes + 255) & ~(size_t)255;
        return r;
    };
    const size_t NFMAX = (size_t)Nn * 1024;
    _Float16* hA   = (_Float16*)alloc(NFMAX * 2);
    _Float16* hB   = (_Float16*)alloc(NFMAX * 2);
    float*    xl   = (float*)alloc(NFMAX * 4);
    float*    xr   = (float*)alloc(NFMAX * 4);
    float*    agg  = (float*)alloc(NFMAX * 4);
    _Float16* wtl  = (_Float16*)alloc((size_t)1024 * 1024 * 2);
    _Float16* wtr  = (_Float16*)alloc((size_t)1024 * 1024 * 2);
    float*    elog = (float*)alloc((size_t)E2 * 4);
    float*    ez   = (float*)alloc((size_t)E2 * 4);
    float*    emax = (float*)alloc((size_t)Nn * 4);
    float*    den  = (float*)alloc((size_t)Nn * 4);

    // h0 = f16(x)
    {
        int nx = Nn * 128;
        gat_f32_to_f16<<<(nx + 255) / 256, 256, 0, stream>>>(x, hA, nx);
    }

    _Float16* hin  = hA;
    _Float16* hout = hB;
    const int waveBlocks = (E2 + 7) / 8;  // 8 waves of 32 per 256-thread block

    for (int L = 0; L < 3; ++L) {
        int Fin = Fd[L], Fout = Fd[L + 1];
        int nw = Fin * Fout;
        gat_convT<<<(nw + 255) / 256, 256, 0, stream>>>(Wl[L], wtl, Fin, Fout);
        gat_convT<<<(nw + 255) / 256, 256, 0, stream>>>(Wr[L], wtr, Fin, Fout);

        dim3 gg((Nn + TM - 1) / TM, Fout / TN);
        gat_gemm_wmma<<<gg, 256, 0, stream>>>(hin, wtl, bl[L], xl, Nn, Fout, Fin);
        gat_gemm_wmma<<<gg, 256, 0, stream>>>(hin, wtr, br[L], xr, Nn, Fout, Fin);

        gat_fill_f32<<<(Nn + 255) / 256, 256, 0, stream>>>(emax, -__builtin_inff(), Nn);
        gat_fill_f32<<<(Nn + 255) / 256, 256, 0, stream>>>(den, 0.0f, Nn);
        int nagg = Nn * Fout;
        gat_fill_f32<<<(nagg + 255) / 256, 256, 0, stream>>>(agg, 0.0f, nagg);

        gat_edge_logit<<<waveBlocks, 256, 0, stream>>>(xl, xr, att[L], srcA, dstA,
                                                       E, Nn, Fout, elog, emax);
        gat_edge_exp<<<(E2 + 255) / 256, 256, 0, stream>>>(elog, emax, dstA, E, Nn, ez, den);
        gat_edge_aggr<<<waveBlocks, 256, 0, stream>>>(ez, den, xl, srcA, dstA,
                                                      E, Nn, Fout, agg);
        gat_finalize<<<(nagg + 255) / 256, 256, 0, stream>>>(agg, bb[L], hout, Nn, Fout);

        _Float16* t = hin; hin = hout; hout = t;
    }

    // classifier: d_out[Nn,64] = h3 @ Wc + bc  (K=128, N=64)
    gat_convT<<<(128 * 64 + 255) / 256, 256, 0, stream>>>(Wc, wtl, 128, 64);
    dim3 gc((Nn + TM - 1) / TM, 1);
    gat_gemm_wmma<<<gc, 256, 0, stream>>>(hin, wtl, bc, (float*)d_out, Nn, 64, 128);
}
